// LRU_5111011082295
// MI455X (gfx1250) — matrix operand order repaired and checked by reading, not compiled, not run on hardware
//
#include <hip/hip_runtime.h>

typedef _Float16 v16h __attribute__((ext_vector_type(16)));
typedef _Float16 v8h  __attribute__((ext_vector_type(8)));
typedef float    v8f  __attribute__((ext_vector_type(8)));
typedef float    v4f  __attribute__((ext_vector_type(4)));
typedef int      v4i  __attribute__((ext_vector_type(4)));

#define HDIM 128   // hidden size (K of big GEMM)
#define IDIM 64    // input size
#define ODIM 32    // output size
#define BT   512   // batch
#define NTC  8     // nTiles processed per block (per blockIdx.y)

// ---------------------------------------------------------------------------
// Async 16B global->LDS copy, with sync fallback if the builtin is absent.
// The builtin's pointer params are int4* in AS1 (global) / AS3 (LDS); casts
// go through uintptr_t (generic LDS addresses carry the LDS byte offset in
// the low 32 bits per the aperture rule).
// ---------------------------------------------------------------------------
typedef __attribute__((address_space(1))) v4i* g4p_t;
typedef __attribute__((address_space(3))) v4i* l4p_t;

__device__ __forceinline__ void copy16_g2l(const _Float16* g, _Float16* l) {
#if __has_builtin(__builtin_amdgcn_global_load_async_to_lds_b128)
    __builtin_amdgcn_global_load_async_to_lds_b128(
        (g4p_t)(uintptr_t)g, (l4p_t)(uint32_t)(uintptr_t)l, 0, 0);
#else
    *(v8h*)l = *(const v8h*)g;   // sync fallback (ds_store_b128)
#endif
}

__device__ __forceinline__ void wait_async0() {
#if __has_builtin(__builtin_amdgcn_s_wait_asynccnt)
    __builtin_amdgcn_s_wait_asynccnt(0);
#else
    asm volatile("s_wait_asynccnt 0x0" ::: "memory");
#endif
}

// ---------------------------------------------------------------------------
// Kernel 1: u[b,h] = gamma[h] * (x0[b,:] . B[h,:])     (fp32, exact)
// ---------------------------------------------------------------------------
__global__ void lru_u_kernel(const float* __restrict__ x0,
                             const float* __restrict__ log_r,
                             const float* __restrict__ Bmat,
                             float* __restrict__ u) {
    const int b = blockIdx.x;
    const int h = threadIdx.x;
    float r = 1.0f / (1.0f + expf(-log_r[h]));
    float gamma = sqrtf(1.0f - r * r + 1e-8f);
    const float* xr = x0 + b * IDIM;
    const float* br = Bmat + h * IDIM;
    float dot = 0.0f;
#pragma unroll
    for (int i = 0; i < IDIM; ++i) dot += xr[i] * br[i];
    u[b * HDIM + h] = gamma * dot;
}

// ---------------------------------------------------------------------------
// Kernel 2: Pt[t,h] = Re(lambda[h]^t) = r^t * cos(t*theta), stored f16
// ---------------------------------------------------------------------------
__global__ void lru_P_kernel(const float* __restrict__ log_r,
                             const float* __restrict__ theta,
                             _Float16* __restrict__ Pt) {
    const int t = blockIdx.x;
    const int h = threadIdx.x;
    float r = 1.0f / (1.0f + expf(-log_r[h]));
    float mag = expf((float)t * logf(r));          // r^t (underflow to 0 is fine)
    float re = mag * cosf((float)t * theta[h]);
    Pt[t * HDIM + h] = (_Float16)re;
}

// ---------------------------------------------------------------------------
// Kernel 3: big WMMA GEMM  out[b,t,o] = sum_h (u[b,h]*C[o,h]) * Pt[t,h]
// M = b*32+o (16384), N = t (T), K = h (128).
// Block = 8 waves; wave w owns mTile = blockIdx.x*8 + w (A fragment in regs,
// built once). The block loops over NTC consecutive nTiles; each 4 KB B-tile
// is async-copied into LDS once (double buffered) and consumed by all 8
// waves via ds_load_b128 -> 8x less L2 read traffic than per-wave loads.
// ---------------------------------------------------------------------------
__global__ __launch_bounds__(256) void
lru_gemm_kernel(const _Float16* __restrict__ Pt,
                const float* __restrict__ u,
                const float* __restrict__ Cm,
                float* __restrict__ out,
                int T, int nTiles) {
    __shared__ _Float16 Bs[2][16 * HDIM];   // 2 x 4 KB

    const int tid   = threadIdx.x;
    const int lane  = tid & 31;
    const int wave  = __builtin_amdgcn_readfirstlane(tid) >> 5;  // scalar
    const int l16   = lane & 15;
    const int lhalf = lane >> 4;

    const int mTile = blockIdx.x * 8 + wave;   // 0 .. 1023
    const int b     = mTile >> 1;
    const int obase = (mTile & 1) << 4;

    // ---- A fragments: A[m,k] = u[b,k]*C[o,k]; ISA 16-bit A 16x32 layout:
    // lanes<16 K-base 0, lanes>=16 K-base 8; V0..V3 = K kb..kb+7,
    // V4..V7 = K kb+16..kb+23.
    const float* up = u + b * HDIM;
    const float* cp = Cm + (obase + l16) * HDIM;
    const int kbA = lhalf * 8;

    v16h A[4];
#pragma unroll
    for (int kk = 0; kk < 4; ++kk) {
        const int k0 = kk * 32 + kbA;
        const int k1 = k0 + 16;
        v16h a;
#pragma unroll
        for (int e = 0; e < 8; ++e) a[e]     = (_Float16)(up[k0 + e] * cp[k0 + e]);
#pragma unroll
        for (int e = 0; e < 8; ++e) a[8 + e] = (_Float16)(up[k1 + e] * cp[k1 + e]);
        A[kk] = a;
    }

    // ---- Cooperative B-tile loader: 256 threads x 16 B = 4 KB tile.
    const int lrow = tid >> 4;          // 0..15  (t row within tile)
    const int lcol = (tid & 15) * 8;    // f16 column offset
    const int nt0  = blockIdx.y * NTC;

    // Prologue: start tile nt0 into buffer 0.
    if (nt0 < nTiles)
        copy16_g2l(Pt + (size_t)(nt0 * 16 + lrow) * HDIM + lcol,
                   &Bs[0][lrow * HDIM + lcol]);

    for (int i = 0; i < NTC; ++i) {
        const int nTile = nt0 + i;
        if (nTile >= nTiles) break;     // block-uniform

        wait_async0();                  // own slice of buf[i&1] landed
        __syncthreads();                // whole tile visible; buf[(i+1)&1] free

        if (i + 1 < NTC && nTile + 1 < nTiles)
            copy16_g2l(Pt + (size_t)((nTile + 1) * 16 + lrow) * HDIM + lcol,
                       &Bs[(i + 1) & 1][lrow * HDIM + lcol]);

        // B fragment from LDS: 32x16, N = lane&15 -> t, K-base = lhalf*8,
        // K kb..kb+7 and kb+16..kb+23 of row Bs[t, :] per lane.
        const _Float16* pB = &Bs[i & 1][l16 * HDIM + lhalf * 8];

        v8f acc = {0.f, 0.f, 0.f, 0.f, 0.f, 0.f, 0.f, 0.f};
#pragma unroll
        for (int kk = 0; kk < 4; ++kk) {
            v8h blo = *(const v8h*)(pB + kk * 32);
            v8h bhi = *(const v8h*)(pB + kk * 32 + 16);
            v16h bb = __builtin_shufflevector(blo, bhi,
                0, 1, 2, 3, 4, 5, 6, 7, 8, 9, 10, 11, 12, 13, 14, 15);
            acc = __builtin_amdgcn_wmma_f32_16x16x32_f16(
                false, A[kk], false, bb, (short)0, acc, false, false);
        }

        // D layout: VGPR j -> o = obase + lhalf*8 + j, N = l16 -> t.
        const int t = nTile * 16 + l16;
        float* op = out + (size_t)b * T * ODIM + (size_t)t * ODIM
                        + obase + lhalf * 8;
        v4f s0 = {acc[0], acc[1], acc[2], acc[3]};
        v4f s1 = {acc[4], acc[5], acc[6], acc[7]};
        *(v4f*)(op)     = s0;
        *(v4f*)(op + 4) = s1;
    }
}

// ---------------------------------------------------------------------------
// Kernel 4: out[:,0,:] += x0 @ (D+F)^T   (fp32, tiny)
// ---------------------------------------------------------------------------
__global__ void lru_out0_kernel(const float* __restrict__ x0,
                                const float* __restrict__ Dm,
                                const float* __restrict__ Fm,
                                float* __restrict__ out, int T) {
    const int idx = blockIdx.x * blockDim.x + threadIdx.x;  // 0..16383
    const int b = idx >> 5;
    const int o = idx & 31;
    const float* xr = x0 + b * IDIM;
    float s = 0.0f;
#pragma unroll
    for (int i = 0; i < IDIM; ++i) s += xr[i] * (Dm[o * IDIM + i] + Fm[o * IDIM + i]);
    out[(size_t)b * T * ODIM + o] += s;
}

// ---------------------------------------------------------------------------
// Kernel 5: final_hidden[b,h,:] = (Re, Im)(lambda[h]^(T-1)) * u[b,h]
// ---------------------------------------------------------------------------
__global__ void lru_fh_kernel(const float* __restrict__ log_r,
                              const float* __restrict__ theta,
                              const float* __restrict__ u,
                              float* __restrict__ fh, int T) {
    const int idx = blockIdx.x * blockDim.x + threadIdx.x;  // 0..65535
    const int b = idx >> 7;
    const int h = idx & 127;
    float r = 1.0f / (1.0f + expf(-log_r[h]));
    float tm1 = (float)(T - 1);
    float mag = expf(tm1 * logf(r));
    float ang = tm1 * theta[h];
    float uu = u[b * HDIM + h];
    fh[(size_t)b * HDIM * 2 + h * 2 + 0] = mag * cosf(ang) * uu;
    fh[(size_t)b * HDIM * 2 + h * 2 + 1] = mag * sinf(ang) * uu;
}

// ---------------------------------------------------------------------------
extern "C" void kernel_launch(void* const* d_in, const int* in_sizes, int n_in,
                              void* d_out, int out_size, void* d_ws, size_t ws_size,
                              hipStream_t stream) {
    const float* x0    = (const float*)d_in[0];
    const float* log_r = (const float*)d_in[1];
    const float* theta = (const float*)d_in[2];
    const float* Bm    = (const float*)d_in[3];
    const float* Cm    = (const float*)d_in[4];
    const float* Dm    = (const float*)d_in[5];
    const float* Fm    = (const float*)d_in[6];
    // d_in[7] is seq_len on device; derive T on host from out_size instead:
    // out_size = BT*T*ODIM + BT*HDIM*2
    const int T = (out_size - BT * HDIM * 2) / (BT * ODIM);

    float* out = (float*)d_out;
    float* fh  = out + (size_t)BT * T * ODIM;

    // workspace: u (fp32, 256 KB) then Pt (f16, T*256 B)
    float*    u  = (float*)d_ws;
    _Float16* Pt = (_Float16*)((char*)d_ws + (size_t)BT * HDIM * sizeof(float));

    lru_u_kernel<<<BT, HDIM, 0, stream>>>(x0, log_r, Bm, u);
    lru_P_kernel<<<T, HDIM, 0, stream>>>(log_r, theta, Pt);

    const int nTiles = T / 16;                            // T=2048 -> 128
    dim3 grid((BT * ODIM) / (16 * 8),                     // 128 mGroups
              (nTiles + NTC - 1) / NTC);                  // 16 nChunks
    lru_gemm_kernel<<<grid, 256, 0, stream>>>(Pt, u, Cm, out, T, nTiles);

    lru_out0_kernel<<<(BT * ODIM) / 256, 256, 0, stream>>>(x0, Dm, Fm, out, T);
    lru_fh_kernel<<<(BT * HDIM) / 256, 256, 0, stream>>>(log_r, theta, u, fh, T);
}
